// TF_53876069761034
// MI455X (gfx1250) — matrix operand order repaired and checked by reading, not compiled, not run on hardware
//
#include <hip/hip_runtime.h>
#include <hip/hip_bf16.h>
#include <stdint.h>

// ---------------------------------------------------------------------------
// Problem constants (from reference): B=16384, M=3, D=2048, DH=1024, DV=512
// out = fc( (softmax(q k^T) col-avg) @ v ),  q=xWq^T, k=xWk^T, v=xWv^T
// ---------------------------------------------------------------------------
#define D_IN        2048
#define KTILES      64            // D / 32
#define NTILES_TOT  160           // (DH + DH + DV) / 16
#define TB          16            // batches per block -> 48 rows = 3 M-tiles

typedef __attribute__((ext_vector_type(16))) __bf16 bf16x16;
typedef __attribute__((ext_vector_type(8)))  __bf16 bf16x8;
typedef __attribute__((ext_vector_type(8)))  float  f32x8;
typedef __attribute__((ext_vector_type(4)))  float  f32x4;   // clang vector: OK
                                                             // for NT builtins

__device__ __forceinline__ unsigned f2bf(float f) {
  union { float f; unsigned u; } c; c.f = f;
  unsigned r = c.u + 0x7FFFu + ((c.u >> 16) & 1u);   // round-to-nearest-even
  return r >> 16;
}
__device__ __forceinline__ float bf2f(unsigned short h) {
  union { unsigned u; float f; } c; c.u = ((unsigned)h) << 16;
  return c.f;
}

__device__ __forceinline__ void wave_lds_fence() {
  // Order per-wave LDS scratch stores before cross-lane LDS reads.
  __builtin_amdgcn_wave_barrier();
  asm volatile("s_wait_dscnt 0" ::: "memory");
  __builtin_amdgcn_wave_barrier();
}

// ---------------------------------------------------------------------------
// Kernel 1: pack Wq|Wk|Wv (f32, [out,in] row-major) into bf16 WMMA B-fragment
// layout: chunk id = ((gnt*KTILES + kt)*32 + lane), 16 bf16 per lane chunk.
// Lane holds column N = nt*16 + (lane&15), K = kt*32 + (lane>>4)*16 + i.
// Each B fragment is then a single contiguous 32B/lane load in the GEMM.
// ---------------------------------------------------------------------------
__global__ void __launch_bounds__(256) pack_weights(
    const float* __restrict__ Wq, const float* __restrict__ Wk,
    const float* __restrict__ Wv, unsigned short* __restrict__ wp) {
  int chunk = blockIdx.x * 256 + threadIdx.x;          // [0, 160*64*32)
  if (chunk >= NTILES_TOT * KTILES * 32) return;
  int lane = chunk & 31;
  int kt   = (chunk >> 5) & (KTILES - 1);
  int gnt  = chunk >> 11;                              // /(32*64)
  int nl = lane & 15, kh = lane >> 4;
  const float* W; int n;
  if (gnt < 64)       { W = Wq; n =  gnt        * 16 + nl; }
  else if (gnt < 128) { W = Wk; n = (gnt - 64)  * 16 + nl; }
  else                { W = Wv; n = (gnt - 128) * 16 + nl; }
  const float* src = W + (size_t)n * D_IN + kt * 32 + kh * 16;
  unsigned short* dst = wp + (size_t)chunk * 16;
  #pragma unroll
  for (int i = 0; i < 16; ++i) dst[i] = (unsigned short)f2bf(src[i]);
}

// ---------------------------------------------------------------------------
// Kernel 2: fused projections + attention + output. One block = 16 batches.
// LDS layout (dynamic, ~308KB):
//   x tile : k-tile blocked; block kt = 48 rows * 64B (32 bf16), 16B chunks
//            swizzled by ((chunk + (row>>2)) & 3) for conflict-free b128 reads.
//            All A-frag addresses within a k-step fit DS 16-bit imm offsets.
//   v tile : 48 rows * 520 bf16
//   q/k wave scratch : 8 waves * 2 * (48*20) f32 (16B-aligned rows)
//   qk logits: 16 batches * 9 f32 ; attn weights: 48 f32
// ---------------------------------------------------------------------------
#define XKBLK    3072                 // 48 rows * 64B per k-tile block
#define X_OFF    0                    // 64 * 3072 = 196608
#define V_OFF    196608
#define VROW_E   520
#define QS_OFF   246528
#define QS_ROWF  20                   // floats per scratch row (80B, aligned)
#define QS_WAVE  7680                 // bytes: 2 * 48 * 20 * 4
#define QK_OFF   307968
#define AW_OFF   308544
#define SMEM_BYTES 308736

__device__ __forceinline__ f32x8 wmma_bf16(bf16x16 a, bf16x16 b, f32x8 c) {
  return __builtin_amdgcn_wmma_f32_16x16x32_bf16(
      /*neg_a=*/false, a, /*neg_b=*/false, b,
      /*c_mod=*/(short)0, c, /*reuse_a=*/false, /*reuse_b=*/false);
}

// A layout (16-bit, 16x32): lanes 0-15 hold K=[0..15], lanes 16-31
// hold K=[16..31] (matches B pack); row M = lane&15. Two swizzled 16B LDS chunks.
__device__ __forceinline__ bf16x16 load_a_frag(const char* smem, int off,
                                               int baseA, int baseB) {
  bf16x8 lo = *(const bf16x8*)(smem + X_OFF + off + baseA);
  bf16x8 hi = *(const bf16x8*)(smem + X_OFF + off + baseB);
  return __builtin_shufflevector(lo, hi, 0, 1, 2, 3, 4, 5, 6, 7,
                                         8, 9, 10, 11, 12, 13, 14, 15);
}

__global__ void __launch_bounds__(256, 1) fused_attn(
    const float* __restrict__ x, const unsigned short* __restrict__ wp,
    const float* __restrict__ Wfc, float* __restrict__ out) {
  extern __shared__ char smem[];
  const int tid  = threadIdx.x;
  const int wave = tid >> 5, lane = tid & 31;
  const int lm = lane & 15, lh = lane >> 4;
  const int bq = blockIdx.x;                       // batch group, 16 batches

  // Per-lane swizzled A-fragment chunk bases (constant across kt/mt since
  // (row>>2)&3 == (lm>>2) for row = mt*16+lm).
  const int sw    = lm >> 2;
  const int baseA = lm * 64 + (((2 * lh)     + sw) & 3) * 16;
  const int baseB = lm * 64 + (((2 * lh + 1) + sw) & 3) * 16;

  // ---- Phase 0: stage x (48 x 2048) into LDS as bf16 (swizzled, NT load) --
  {
    const f32x4* xg = (const f32x4*)(x + (size_t)bq * 48 * D_IN);
    for (int idx = tid; idx < 48 * 256; idx += 256) {  // 16B chunks
      int row = idx >> 8;
      int cc  = idx & 255;                             // chunk within row
      f32x4 f0 = __builtin_nontemporal_load(xg + row * 512 + cc * 2);
      f32x4 f1 = __builtin_nontemporal_load(xg + row * 512 + cc * 2 + 1);
      uint4 u;
      u.x = f2bf(f0.x) | (f2bf(f0.y) << 16);
      u.y = f2bf(f0.z) | (f2bf(f0.w) << 16);
      u.z = f2bf(f1.x) | (f2bf(f1.y) << 16);
      u.w = f2bf(f1.z) | (f2bf(f1.w) << 16);
      int kt = cc >> 2, c = cc & 3;
      int cs = (c + ((row >> 2) & 3)) & 3;             // bank swizzle
      *(uint4*)(smem + X_OFF + kt * XKBLK + row * 64 + cs * 16) = u;
    }
    if (tid < 144) ((float*)(smem + QK_OFF))[tid] = 0.0f;
  }
  __syncthreads();

  // ---- Phase 1: WMMA GEMMs. 96 tasks over 8 waves:
  //   t in [0,64): paired q/k h-tile t  -> logits qk[b][i][j] accumulation
  //   t in [64,96): v n-tile (t-64)     -> v tile into LDS (bf16)
  const bf16x16* WP = (const bf16x16*)wp;
  float* qk = (float*)(smem + QK_OFF);
  for (int t = wave; t < 96; t += 8) {
    if (t < 64) {
      const size_t baseq = ((size_t)t        * KTILES) * 32 + lane;
      const size_t basek = ((size_t)(t + 64) * KTILES) * 32 + lane;
      f32x8 aq[3] = {}, ak[3] = {};
      for (int kt = 0; kt < KTILES; ++kt) {
        int kp = (kt + 8 < KTILES) ? kt + 8 : KTILES - 1;
        __builtin_prefetch(&WP[baseq + (size_t)kp * 32], 0, 3);
        __builtin_prefetch(&WP[basek + (size_t)kp * 32], 0, 3);
        bf16x16 fbq = WP[baseq + (size_t)kt * 32];
        bf16x16 fbk = WP[basek + (size_t)kt * 32];
        const int koff = kt * XKBLK;
        #pragma unroll
        for (int mt = 0; mt < 3; ++mt) {
          bf16x16 a = load_a_frag(smem, koff + mt * 1024, baseA, baseB);
          aq[mt] = wmma_bf16(a, fbq, aq[mt]);
          ak[mt] = wmma_bf16(a, fbk, ak[mt]);
        }
      }
      // Spill q/k tiles (48x16 each) to this wave's LDS scratch.
      // C layout: VGPR r -> row M = mt*16 + (lane<16 ? r : r+8), col N = lane&15.
      float* qsc = (float*)(smem + QS_OFF + wave * QS_WAVE);
      float* ksc = qsc + 48 * QS_ROWF;
      #pragma unroll
      for (int mt = 0; mt < 3; ++mt)
        #pragma unroll
        for (int r = 0; r < 8; ++r) {
          int row = mt * 16 + (lh ? r + 8 : r);
          qsc[row * QS_ROWF + lm] = aq[mt][r];
          ksc[row * QS_ROWF + lm] = ak[mt][r];
        }
      wave_lds_fence();
      // Contract this 16-wide h slab into per-batch 3x3 logits (f32 atomics).
      for (int c = lane; c < 144; c += 32) {
        int b = c / 9, rem = c % 9, i = rem / 3, j = rem % 3;
        const float4* qr = (const float4*)(qsc + (3 * b + i) * QS_ROWF);
        const float4* kr = (const float4*)(ksc + (3 * b + j) * QS_ROWF);
        float s = 0.f;
        #pragma unroll
        for (int n = 0; n < 4; ++n) {
          float4 a = qr[n], d = kr[n];
          s += a.x * d.x + a.y * d.y + a.z * d.z + a.w * d.w;
        }
        atomicAdd(&qk[c], s);
      }
    } else {
      const int vt = t - 64;
      const size_t basev = ((size_t)(128 + vt) * KTILES) * 32 + lane;
      f32x8 av[3] = {};
      for (int kt = 0; kt < KTILES; ++kt) {
        int kp = (kt + 8 < KTILES) ? kt + 8 : KTILES - 1;
        __builtin_prefetch(&WP[basev + (size_t)kp * 32], 0, 3);
        bf16x16 fb = WP[basev + (size_t)kt * 32];
        const int koff = kt * XKBLK;
        #pragma unroll
        for (int mt = 0; mt < 3; ++mt) {
          bf16x16 a = load_a_frag(smem, koff + mt * 1024, baseA, baseB);
          av[mt] = wmma_bf16(a, fb, av[mt]);
        }
      }
      unsigned short* vb = (unsigned short*)(smem + V_OFF);
      #pragma unroll
      for (int mt = 0; mt < 3; ++mt)
        #pragma unroll
        for (int r = 0; r < 8; ++r) {
          int row = mt * 16 + (lh ? r + 8 : r);
          vb[row * VROW_E + vt * 16 + lm] = (unsigned short)f2bf(av[mt][r]);
        }
    }
  }
  __syncthreads();

  // ---- Phase 2: softmax over j per (b,i), average over queries i ----
  float* aw = (float*)(smem + AW_OFF);
  if (tid < 16) {
    const float* q9 = qk + tid * 9;
    float w0 = 0.f, w1 = 0.f, w2 = 0.f;
    #pragma unroll
    for (int i = 0; i < 3; ++i) {
      float a0 = q9[i * 3 + 0], a1 = q9[i * 3 + 1], a2 = q9[i * 3 + 2];
      float m  = fmaxf(a0, fmaxf(a1, a2));
      float e0 = __expf(a0 - m), e1 = __expf(a1 - m), e2 = __expf(a2 - m);
      float inv = 1.f / (e0 + e1 + e2);
      w0 += e0 * inv; w1 += e1 * inv; w2 += e2 * inv;
    }
    const float third = (1.f / 3.f);
    aw[tid * 3 + 0] = w0 * third;
    aw[tid * 3 + 1] = w1 * third;
    aw[tid * 3 + 2] = w2 * third;
  }
  __syncthreads();

  // ---- Phase 3: out[b] = sum_c (sum_j aw[b][j] * v[3b+j][c]) * Wfc[c] ----
  {
    int b  = tid >> 4;                      // 0..15
    int cs = tid & 15;                      // 32 columns each
    float w0 = aw[b * 3 + 0], w1 = aw[b * 3 + 1], w2 = aw[b * 3 + 2];
    const unsigned short* vb = (const unsigned short*)(smem + V_OFF);
    const unsigned short* v0 = vb + (size_t)(3 * b + 0) * VROW_E;
    const unsigned short* v1 = vb + (size_t)(3 * b + 1) * VROW_E;
    const unsigned short* v2 = vb + (size_t)(3 * b + 2) * VROW_E;
    float acc = 0.f;
    #pragma unroll 4
    for (int c = cs * 32; c < cs * 32 + 32; ++c) {
      float vv = w0 * bf2f(v0[c]) + w1 * bf2f(v1[c]) + w2 * bf2f(v2[c]);
      acc += vv * Wfc[c];
    }
    acc += __shfl_xor(acc, 8, 32);          // reduce within 16-lane half
    acc += __shfl_xor(acc, 4, 32);
    acc += __shfl_xor(acc, 2, 32);
    acc += __shfl_xor(acc, 1, 32);
    if (cs == 0) out[bq * TB + b] = acc;
  }
}

// ---------------------------------------------------------------------------
extern "C" void kernel_launch(void* const* d_in, const int* in_sizes, int n_in,
                              void* d_out, int out_size, void* d_ws,
                              size_t ws_size, hipStream_t stream) {
  (void)in_sizes; (void)n_in; (void)out_size; (void)ws_size;
  const float* x   = (const float*)d_in[0];
  const float* Wq  = (const float*)d_in[1];
  const float* Wk  = (const float*)d_in[2];
  const float* Wv  = (const float*)d_in[3];
  const float* Wfc = (const float*)d_in[4];
  unsigned short* wp = (unsigned short*)d_ws;   // 10.5 MB bf16 packed weights

  (void)hipFuncSetAttribute((const void*)fused_attn,
                            hipFuncAttributeMaxDynamicSharedMemorySize,
                            SMEM_BYTES);

  pack_weights<<<(NTILES_TOT * KTILES * 32) / 256, 256, 0, stream>>>(
      Wq, Wk, Wv, wp);
  fused_attn<<<16384 / TB, 256, SMEM_BYTES, stream>>>(
      x, wp, Wfc, (float*)d_out);
}
